// MeanAveragePrecision_87230785781780
// MI455X (gfx1250) — matrix operand order repaired and checked by reading, not compiled, not run on hardware
//
#include <hip/hip_runtime.h>
#include <hip/hip_bf16.h>

#define NUM_CLASSES 80
#define N_IMAGES    16
#define D_N         8192
#define G_N         4096
#define EPS_F       1e-6f
#define IOU_TH      0.5f
#define MAX_BGT     256

typedef __attribute__((ext_vector_type(16))) _Float16 v16h;
typedef __attribute__((ext_vector_type(8)))  float    v8f;

// ---------------------------------------------------------------------------
// Async global->LDS staging (CDNA5 GLOBAL_LOAD_ASYNC_TO_LDS, ASYNCcnt path).
// GVS addressing: mem = SADDR(64b sgpr) + VADDR(32b vgpr) + IOFFSET.
// vdst VGPR carries the LDS byte offset.
// ---------------------------------------------------------------------------
__device__ __forceinline__ void async_g2l_b128(unsigned lds_off, unsigned g_off,
                                               const void* gbase) {
  asm volatile("global_load_async_to_lds_b128 %0, %1, %2 offset:0"
               :
               : "v"(lds_off), "v"(g_off),
                 "s"((unsigned long long)(uintptr_t)gbase)
               : "memory");
}
__device__ __forceinline__ void async_wait0() {
  asm volatile("s_wait_asynccnt 0x0" ::: "memory");
}
__device__ __forceinline__ unsigned lds_offset_of(const void* p) {
  return (unsigned)(uintptr_t)p;   // addrspace(3) pointers are LDS byte offsets
}

// ---------------------------------------------------------------------------
// Kernel 1: bitonic sort of detections by (class asc, score desc, idx asc)
// 64-bit keys live entirely in LDS (64KB of the 320KB WGP LDS).
// Also produces per-class start offsets (exclusive scan of class histogram).
// ---------------------------------------------------------------------------
__global__ __launch_bounds__(1024) void map_sort_kernel(
    const float* __restrict__ det_scores, const int* __restrict__ det_labels,
    int* __restrict__ order, int* __restrict__ class_start) {
  __shared__ unsigned long long keys[D_N];   // 64 KB
  __shared__ int cnt[NUM_CLASSES];
  const int tid = threadIdx.x;

  for (int i = tid; i < D_N; i += 1024) {
    unsigned int b = __float_as_uint(det_scores[i]);
    unsigned int ord  = (b & 0x80000000u) ? ~b : (b | 0x80000000u);
    unsigned int desc = ~ord;
    unsigned long long key =
        ((unsigned long long)(det_labels[i] & 0x7F) << 45) |
        ((unsigned long long)desc << 13) |
        (unsigned long long)(i & 0x1FFF);
    keys[i] = key;
  }
  __syncthreads();

  for (int k = 2; k <= D_N; k <<= 1) {
    for (int j = k >> 1; j > 0; j >>= 1) {
      for (int i = tid; i < D_N; i += 1024) {
        int ixj = i ^ j;
        if (ixj > i) {
          bool asc = ((i & k) == 0);
          unsigned long long a = keys[i], b2 = keys[ixj];
          if ((a > b2) == asc) { keys[i] = b2; keys[ixj] = a; }
        }
      }
      __syncthreads();
    }
  }

  for (int c = tid; c < NUM_CLASSES; c += 1024) cnt[c] = 0;
  __syncthreads();
  for (int i = tid; i < D_N; i += 1024) {
    unsigned long long key = keys[i];
    order[i] = (int)(key & 0x1FFF);
    atomicAdd(&cnt[(int)(key >> 45)], 1);
  }
  __syncthreads();
  if (tid == 0) {
    int acc = 0;
    for (int c = 0; c < NUM_CLASSES; ++c) { class_start[c] = acc; acc += cnt[c]; }
    class_start[NUM_CLASSES] = acc;   // == D_N
  }
}

// ---------------------------------------------------------------------------
// Kernel 2: bucket GT counts via WMMA.
//   counts[c][img] = onehot_label(80xG) @ onehot_img(Gx16)
// Exact 0/1 f16 operands, f32 accumulate. One wave per 16-class tile.
// gt_labels / gt_image_ids staged to LDS with async loads first, so the
// WMMA operand construction reads ds_load_b32 instead of scattered vmem.
// ---------------------------------------------------------------------------
__global__ __launch_bounds__(32) void map_bucket_count_kernel(
    const int* __restrict__ gt_labels, const int* __restrict__ gt_ids,
    float* __restrict__ counts) {
  __shared__ int s_lab[G_N];   // 16 KB
  __shared__ int s_img[G_N];   // 16 KB
  const int tile = blockIdx.x;          // classes [tile*16, tile*16+16)
  const int lane = threadIdx.x;
  const int grp  = lane >> 4;           // lane group 0 / 1
  const int M    = lane & 15;           // row (A) / col (B,D) index

  // stage both arrays into LDS: 16 KB each, 16 B per lane per issue
  {
    unsigned lab0 = lds_offset_of(&s_lab[0]);
    unsigned img0 = lds_offset_of(&s_img[0]);
    for (unsigned off = (unsigned)lane * 16u; off < G_N * 4u; off += 32u * 16u) {
      async_g2l_b128(lab0 + off, off, gt_labels);
      async_g2l_b128(img0 + off, off, gt_ids);
    }
    async_wait0();
  }
  __syncthreads();

  v8f acc = {};
  for (int g0 = 0; g0 < G_N; g0 += 32) {
    v16h a = {}, b = {};
#pragma unroll
    for (int e = 0; e < 16; ++e) {
      // A (16x32 f16): lane group selects K half, K = e + grp*16
      int Ka = e + grp * 16;
      a[e] = (s_lab[g0 + Ka] == tile * 16 + M) ? (_Float16)1.0f
                                               : (_Float16)0.0f;
      // B (32x16 f16): lane group selects K half, N = lane&15
      int Kb = e + grp * 16;
      b[e] = (s_img[g0 + Kb] == M) ? (_Float16)1.0f : (_Float16)0.0f;
    }
    acc = __builtin_amdgcn_wmma_f32_16x16x32_f16(
        /*neg_a=*/false, a, /*neg_b=*/false, b,
        /*c_mod=*/(short)0, acc, /*reuse_a=*/false, /*reuse_b=*/false);
  }
  // D (16x16 f32): VGPR v -> row v + grp*8, col = lane&15
#pragma unroll
  for (int v = 0; v < 8; ++v) {
    int row = v + grp * 8;
    counts[(tile * 16 + row) * N_IMAGES + M] = acc[v];
  }
}

// ---------------------------------------------------------------------------
// Kernel 3: greedy matching per (class, image) bucket. One wave32 per bucket.
// gt labels/ids staged to LDS via async loads (kills 1280x re-reads of the
// same 32 KB from L2). GTs compacted in ascending global index so the
// lowest-index tie break reproduces jnp.argmax first-tie semantics.
// ---------------------------------------------------------------------------
__global__ __launch_bounds__(32) void map_match_kernel(
    const float* __restrict__ det_boxes, const int* __restrict__ det_ids,
    const float* __restrict__ gt_boxes,  const int* __restrict__ gt_labels,
    const int* __restrict__ gt_ids,      const int* __restrict__ order,
    const int* __restrict__ class_start, const float* __restrict__ counts,
    float* __restrict__ tp) {
  const int bkt  = blockIdx.x;
  const int c    = bkt >> 4;            // /N_IMAGES
  const int img  = bkt & 15;
  const int lane = threadIdx.x;

  __shared__ int   s_lab[G_N];          // 16 KB
  __shared__ int   s_img[G_N];          // 16 KB
  __shared__ float gx1[MAX_BGT], gy1[MAX_BGT], gx2[MAX_BGT], gy2[MAX_BGT];
  __shared__ float garea[MAX_BGT];
  __shared__ int   matched[MAX_BGT];

  // async-stage the id arrays while we init LDS state
  {
    unsigned lab0 = lds_offset_of(&s_lab[0]);
    unsigned img0 = lds_offset_of(&s_img[0]);
    for (unsigned off = (unsigned)lane * 16u; off < G_N * 4u; off += 32u * 16u) {
      async_g2l_b128(lab0 + off, off, gt_labels);
      async_g2l_b128(img0 + off, off, gt_ids);
    }
  }
  for (int i = lane; i < MAX_BGT; i += 32) matched[i] = 0;

  // consume the WMMA-produced bucket count (skip gather for empty buckets)
  const int cnt_hint = (int)(counts[c * N_IMAGES + img] + 0.5f);

  async_wait0();
  __syncthreads();

  int ngt = 0;
  if (cnt_hint > 0) {
    int base = 0;
    for (int g0 = 0; g0 < G_N; g0 += 32) {
      if (g0 + 32 < G_N) __builtin_prefetch(&gt_boxes[(size_t)(g0 + 32) * 4], 0, 3);
      int g = g0 + lane;
      bool m = (s_lab[g] == c) && (s_img[g] == img);
      unsigned long long mask = __ballot(m);   // wave32: upper bits zero
      int pos = base + (int)__popcll(mask & ((1ull << lane) - 1ull));
      if (m && pos < MAX_BGT) {
        float x1 = gt_boxes[g * 4 + 0], y1 = gt_boxes[g * 4 + 1];
        float x2 = gt_boxes[g * 4 + 2], y2 = gt_boxes[g * 4 + 3];
        gx1[pos] = x1; gy1[pos] = y1; gx2[pos] = x2; gy2[pos] = y2;
        garea[pos] = fabsf((x2 - x1) * (y2 - y1));
      }
      base += (int)__popcll(mask);
    }
    ngt = (base < MAX_BGT) ? base : MAX_BGT;
  }
  __syncthreads();

  const int dstart = class_start[c], dend = class_start[c + 1];
  for (int t = dstart; t < dend; ++t) {
    int d = order[t];
    if (det_ids[d] != img) continue;    // uniform across the wave
    float dx1 = det_boxes[d * 4 + 0], dy1 = det_boxes[d * 4 + 1];
    float dx2 = det_boxes[d * 4 + 2], dy2 = det_boxes[d * 4 + 3];
    float darea = fabsf((dx2 - dx1) * (dy2 - dy1));

    float best_iou = -1.0f; int best_idx = 0x7FFFFFFF;
    for (int g = lane; g < ngt; g += 32) {
      float xx1 = fmaxf(dx1, gx1[g]);
      float yy1 = fmaxf(dy1, gy1[g]);
      float xx2 = fminf(dx2, gx2[g]);
      float yy2 = fminf(dy2, gy2[g]);
      float inter = fmaxf(xx2 - xx1, 0.0f) * fmaxf(yy2 - yy1, 0.0f);
      float iou = inter / (darea + garea[g] - inter + EPS_F);
      if (iou > best_iou || (iou == best_iou && g < best_idx)) {
        best_iou = iou; best_idx = g;
      }
    }
#pragma unroll
    for (int off = 16; off > 0; off >>= 1) {
      float oi = __shfl_down(best_iou, off, 32);
      int   ox = __shfl_down(best_idx, off, 32);
      if (oi > best_iou || (oi == best_iou && ox < best_idx)) {
        best_iou = oi; best_idx = ox;
      }
    }
    if (lane == 0) {
      int is_tp = 0;
      if (ngt > 0 && best_iou > IOU_TH && matched[best_idx] == 0) {
        matched[best_idx] = 1; is_tp = 1;
      }
      tp[t] = (float)is_tp;
    }
  }
}

// ---------------------------------------------------------------------------
// Kernel 4: per-class trapezoid AP + mean. One thread per class.
// Within a class's contiguous sorted range, denom = j+1, so
// prec = tpc/(j+1+eps); initial (r=0, p=1) matches the prepended point and
// out-of-class positions contribute zero area.
// ---------------------------------------------------------------------------
__global__ __launch_bounds__(128) void map_ap_kernel(
    const float* __restrict__ counts, const int* __restrict__ class_start,
    const float* __restrict__ tp, float* __restrict__ out) {
  __shared__ float ap_s[NUM_CLASSES];
  __shared__ float hg_s[NUM_CLASSES];
  const int c = threadIdx.x;
  if (c < NUM_CLASSES) {
    float ngt = 0.0f;
#pragma unroll
    for (int i = 0; i < N_IMAGES; ++i) ngt += counts[c * N_IMAGES + i];
    ngt = floorf(ngt + 0.5f);   // WMMA accumulation is exact integer anyway

    const int s = class_start[c], e = class_start[c + 1];
    float tpc = 0.0f, rprev = 0.0f, pprev = 1.0f, ap = 0.0f;
    for (int j = s; j < e; ++j) {
      tpc += tp[j];
      float rec  = tpc / (ngt + EPS_F);
      float prec = tpc / ((float)(j - s + 1) + EPS_F);
      ap += (rec - rprev) * (prec + pprev) * 0.5f;
      rprev = rec; pprev = prec;
    }
    bool has_gt = (ngt > 0.0f);
    ap_s[c] = has_gt ? ap : 0.0f;
    hg_s[c] = has_gt ? 1.0f : 0.0f;
  }
  __syncthreads();
  if (threadIdx.x == 0) {
    float ssum = 0.0f, n = 0.0f;
    for (int i = 0; i < NUM_CLASSES; ++i) { ssum += ap_s[i]; n += hg_s[i]; }
    out[0] = ssum / fmaxf(n, 1.0f);
  }
}

// ---------------------------------------------------------------------------
extern "C" void kernel_launch(void* const* d_in, const int* in_sizes, int n_in,
                              void* d_out, int out_size, void* d_ws, size_t ws_size,
                              hipStream_t stream) {
  const float* det_boxes  = (const float*)d_in[0];
  const float* det_scores = (const float*)d_in[1];
  const int*   det_labels = (const int*)d_in[2];
  const int*   det_ids    = (const int*)d_in[3];
  const float* gt_boxes   = (const float*)d_in[4];
  const int*   gt_labels  = (const int*)d_in[5];
  const int*   gt_ids     = (const int*)d_in[6];
  (void)in_sizes; (void)n_in; (void)out_size; (void)ws_size;

  char* ws = (char*)d_ws;
  int*   order       = (int*)ws;    ws += (size_t)D_N * sizeof(int);
  int*   class_start = (int*)ws;    ws += 128 * sizeof(int);
  float* counts      = (float*)ws;  ws += (size_t)NUM_CLASSES * N_IMAGES * sizeof(float);
  float* tp          = (float*)ws;  ws += (size_t)D_N * sizeof(float);
  float* out         = (float*)d_out;

  map_sort_kernel<<<1, 1024, 0, stream>>>(det_scores, det_labels, order, class_start);
  map_bucket_count_kernel<<<NUM_CLASSES / 16, 32, 0, stream>>>(gt_labels, gt_ids, counts);
  map_match_kernel<<<NUM_CLASSES * N_IMAGES, 32, 0, stream>>>(
      det_boxes, det_ids, gt_boxes, gt_labels, gt_ids, order, class_start, counts, tp);
  map_ap_kernel<<<1, 128, 0, stream>>>(counts, class_start, tp, out);
}
